// Encoder_1262720385181
// MI455X (gfx1250) — matrix operand order repaired and checked by reading, not compiled, not run on hardware
//
#include <hip/hip_runtime.h>
#include <hip/hip_bf16.h>
#include <math.h>

// ---------------------------------------------------------------------------
// CDNA5 (gfx1250, wave32) WMMA types
// ---------------------------------------------------------------------------
typedef __attribute__((ext_vector_type(16))) _Float16 v16h;
typedef __attribute__((ext_vector_type(8)))  float    v8f;

#define B_   4
#define NP_  16384
#define NR_  2048
#define E_   65536
#define LAT_ 128
#define WAVES_ 4   // waves per block (wave32)

static __device__ __forceinline__ float silu_f(float x) {
    return x / (1.0f + __expf(-x));
}

// ---------------------------------------------------------------------------
// Fragment builders (CDNA5 wave32 layouts, CDNA5 ISA §7.12.2)
// A 16x32 f16: lane = 16*hi + m ; VGPR r holds K pair:
//   k = (r&3)*2 + hi*8 + (r>>2)*16
// B 32x16 f16 (weights stored transposed [N][Kpad]): lane = 16*hi + n ;
//   VGPR r holds K pair: k = (r&3)*2 + hi*8 + (r>>2)*16  (same as A)
// C/D 16x16 f32: VGPR r, lane L -> row m = r + 8*(L>>4), col n = L&15
// ---------------------------------------------------------------------------
template <class F>
static __device__ __forceinline__ v16h build_a_frag(const F& f, int rowbase,
                                                    int kbase, int lane) {
    const int m  = lane & 15;
    const int hi = lane >> 4;
    const int row = rowbase + m;
    v16h a;
#pragma unroll
    for (int r = 0; r < 8; ++r) {
        const int k = kbase + ((r & 3) * 2) + hi * 8 + ((r >> 2) * 16);
        a[2 * r]     = (_Float16)f(row, k);
        a[2 * r + 1] = (_Float16)f(row, k + 1);
    }
    return a;
}

static __device__ __forceinline__ v16h load_b_frag(const _Float16* wt, int kpad,
                                                   int nbase, int kbase, int lane) {
    const int n  = nbase + (lane & 15);
    const int hi = lane >> 4;
    const _Float16* p = wt + (size_t)n * kpad + kbase + hi * 8;
    v16h b;
#pragma unroll
    for (int r = 0; r < 8; ++r) {
        b[2 * r]     = p[(r & 3) * 2 + (r >> 2) * 16];
        b[2 * r + 1] = p[(r & 3) * 2 + (r >> 2) * 16 + 1];
    }
    return b;
}

static __device__ __forceinline__ v16h load_a_frag_lds16(const _Float16* ht,
                                                         int kbase, int lane) {
    const int m  = lane & 15;
    const int hi = lane >> 4;
    const _Float16* p = ht + m * LAT_;
    v16h a;
#pragma unroll
    for (int r = 0; r < 8; ++r) {
        const int k = kbase + ((r & 3) * 2) + hi * 8 + ((r >> 2) * 16);
        a[2 * r]     = p[k];
        a[2 * r + 1] = p[k + 1];
    }
    return a;
}

// A fragment from an f32 LDS stage tile [16][dstride]
static __device__ __forceinline__ v16h load_a_frag_stage(const float* st,
                                                         int dstride, int kbase,
                                                         int lane) {
    const int m  = lane & 15;
    const int hi = lane >> 4;
    const float* p = st + m * dstride;
    v16h a;
#pragma unroll
    for (int r = 0; r < 8; ++r) {
        const int k = kbase + ((r & 3) * 2) + hi * 8 + ((r >> 2) * 16);
        a[2 * r]     = (_Float16)p[k];
        a[2 * r + 1] = (_Float16)p[k + 1];
    }
    return a;
}

// ---------------------------------------------------------------------------
// Async global->LDS staging of NBYTES contiguous bytes per wave
// (gfx1250 GLOBAL_LOAD_ASYNC_TO_LDS_B128, GV mode; tracked by ASYNCcnt).
// Each instruction moves 32 lanes x 16B = 512B.
// ---------------------------------------------------------------------------
template <int NBYTES>
static __device__ __forceinline__ void async_stage(const float* gsrc,
                                                   float* stage, int lane) {
    unsigned ldsa = (unsigned)(uintptr_t)(void*)stage + (unsigned)lane * 16u;
    unsigned long long ga =
        (unsigned long long)(uintptr_t)gsrc + (unsigned long long)lane * 16ull;
#pragma unroll
    for (int it = 0; it < NBYTES / 512; ++it) {
        asm volatile("global_load_async_to_lds_b128 %0, %1, off"
                     :: "v"(ldsa), "v"(ga)
                     : "memory");
        ldsa += 512u;
        ga   += 512ull;
    }
    asm volatile("s_wait_asynccnt 0x0" ::: "memory");
}

// ---------------------------------------------------------------------------
// Generic MLP block tile: y = condLN( silu(x@W0+b0) @ W1 + b1 ), out (+=resid)
// One wave computes a 16-row x 128-col tile with v_wmma_f32_16x16x32_f16.
// DENSE = number of leading A columns that are a contiguous row-major f32
// matrix `dsrc` (async-staged to LDS); remaining columns come from `afetch`.
// ---------------------------------------------------------------------------
template <int KPAD, int DENSE, class F>
static __device__ void mlp_block_tile(const F& afetch, const float* dsrc,
                                      int rowbase,
                                      const _Float16* w0t, const _Float16* w1t,
                                      const float* b0, const float* b1,
                                      const float* cscale,   // 1 + scale [128]
                                      const float* cshift,   // shift     [128]
                                      const float* resid,    // nullptr or out
                                      float* out,
                                      _Float16* htile,       // per-wave [16*128]
                                      float* stage) {        // per-wave [16*DENSE]
    const int lane = threadIdx.x & 31;
    const int nlo  = lane & 15;
    const int hi   = lane >> 4;

    if constexpr (DENSE > 0) {
        async_stage<16 * DENSE * 4>(dsrc + (size_t)rowbase * DENSE, stage, lane);
    }

    v8f acc[8];
#pragma unroll
    for (int nt = 0; nt < 8; ++nt) {
        const float bb = b0[nt * 16 + nlo];
#pragma unroll
        for (int r = 0; r < 8; ++r) acc[nt][r] = bb;
    }

    // ---- GEMM 1: x @ W0 ----
#pragma unroll
    for (int kt = 0; kt < KPAD / 32; ++kt) {
        v16h a;
        if (kt * 32 < DENSE)
            a = load_a_frag_stage(stage, DENSE, kt * 32, lane);
        else
            a = build_a_frag(afetch, rowbase, kt * 32, lane);
#pragma unroll
        for (int nt = 0; nt < 8; ++nt) {
            const v16h b = load_b_frag(w0t, KPAD, nt * 16, kt * 32, lane);
            acc[nt] = __builtin_amdgcn_wmma_f32_16x16x32_f16(
                false, a, false, b, (short)0, acc[nt], false, false);
        }
    }

    // ---- silu -> f16 hidden tile in LDS ----
#pragma unroll
    for (int nt = 0; nt < 8; ++nt) {
#pragma unroll
        for (int r = 0; r < 8; ++r) {
            const int m = r + 8 * hi;
            htile[m * LAT_ + nt * 16 + nlo] = (_Float16)silu_f(acc[nt][r]);
        }
    }
    __syncthreads();

    // ---- GEMM 2: h @ W1 ----
#pragma unroll
    for (int nt = 0; nt < 8; ++nt) {
        const float bb = b1[nt * 16 + nlo];
#pragma unroll
        for (int r = 0; r < 8; ++r) acc[nt][r] = bb;
    }
#pragma unroll
    for (int kt = 0; kt < 4; ++kt) {
        const v16h a = load_a_frag_lds16(htile, kt * 32, lane);
#pragma unroll
        for (int nt = 0; nt < 8; ++nt) {
            const v16h b = load_b_frag(w1t, LAT_, nt * 16, kt * 32, lane);
            acc[nt] = __builtin_amdgcn_wmma_f32_16x16x32_f16(
                false, a, false, b, (short)0, acc[nt], false, false);
        }
    }

    // ---- conditional LayerNorm in registers via shfl_xor butterflies ----
    float psum[8], psq[8];
#pragma unroll
    for (int r = 0; r < 8; ++r) { psum[r] = 0.f; psq[r] = 0.f; }
#pragma unroll
    for (int nt = 0; nt < 8; ++nt)
#pragma unroll
        for (int r = 0; r < 8; ++r) {
            const float x = acc[nt][r];
            psum[r] += x;
            psq[r]  += x * x;
        }
#pragma unroll
    for (int msk = 1; msk < 16; msk <<= 1) {
#pragma unroll
        for (int r = 0; r < 8; ++r) {
            psum[r] += __shfl_xor(psum[r], msk, 32);
            psq[r]  += __shfl_xor(psq[r],  msk, 32);
        }
    }
    float mean[8], rstd[8];
#pragma unroll
    for (int r = 0; r < 8; ++r) {
        mean[r] = psum[r] * (1.0f / 128.0f);
        const float var = psq[r] * (1.0f / 128.0f) - mean[r] * mean[r];
        rstd[r] = rsqrtf(var + 1e-6f);
    }
#pragma unroll
    for (int nt = 0; nt < 8; ++nt) {
        const int n = nt * 16 + nlo;
        const float sc = cscale[n];
        const float sh = cshift[n];
#pragma unroll
        for (int r = 0; r < 8; ++r) {
            const int row = rowbase + r + 8 * hi;
            const float xn = (acc[nt][r] - mean[r]) * rstd[r];
            float y = xn * sc + sh;
            if (resid) {
                if constexpr (DENSE == LAT_)    // resid == dsrc: reuse LDS stage
                    y += stage[(r + 8 * hi) * LAT_ + n];
                else
                    y += resid[(size_t)row * LAT_ + n];
            }
            out[(size_t)row * LAT_ + n] = y;
        }
    }
}

// ---------------------------------------------------------------------------
// A-fetch functors (handle concat / gather / zero-pad in the A operand)
// ---------------------------------------------------------------------------
struct PnEmbF {              // [feat(64) | pstruct(4) | pad] rows = B*NP
    const float* feat; const float* ps;
    __device__ float operator()(int row, int k) const {
        if (k < 64) return feat[(size_t)row * 64 + k];
        if (k < 68) return ps[(size_t)row * 4 + (k - 64)];
        return 0.f;
    }
};
struct RnEmbF {              // effective input = rstruct(4) (dummy zeros folded)
    const float* rs;
    __device__ float operator()(int row, int k) const {
        return (k < 4) ? rs[(size_t)row * 4 + k] : 0.f;
    }
};
struct EEmbF {               // edge feats (3) + pad
    const float* ef;
    __device__ float operator()(int row, int k) const {
        return (k < 3) ? ef[(size_t)row * 3 + k] : 0.f;
    }
};
struct EUpdF {               // k>=128: [p_lat[sender] | r_lat[recv]], rows = B*E
    const float* elat; const float* plat; const float* rlat;
    const int* send; const int* recv;
    __device__ float operator()(int row, int k) const {
        if (k < 128) return elat[(size_t)row * LAT_ + k];   // unused (staged)
        const int b = row >> 16;                            // E = 65536
        if (k < 256) {
            const int s = send[row];
            return plat[((size_t)(b << 14) + s) * LAT_ + (k - 128)];
        }
        const int r = recv[row];
        return rlat[((size_t)(b << 11) + r) * LAT_ + (k - 256)];
    }
};
struct RnUpdF {              // k>=128: segment_mean(e_lat), rows = B*NR
    const float* rlat; const float* sums; const float* cnts;
    __device__ float operator()(int row, int k) const {
        if (k < 128) return rlat[(size_t)row * LAT_ + k];   // unused (staged)
        const float c = cnts[row];
        return sums[(size_t)row * LAT_ + (k - 128)] * (1.0f / fmaxf(c, 1.0f));
    }
};
struct PnUpdF {              // fully staged; functor unused
    const float* plat;
    __device__ float operator()(int row, int k) const {
        return plat[(size_t)row * LAT_ + k];
    }
};

// ---------------------------------------------------------------------------
// Kernels
// ---------------------------------------------------------------------------
template <int KPAD, int DENSE, class F>
__global__ __launch_bounds__(32 * WAVES_) void mlp_kernel(
        F afetch, const float* dsrc,
        const _Float16* w0t, const _Float16* w1t,
        const float* b0, const float* b1,
        const float* cscale, const float* cshift,
        const float* resid, float* out) {
    __shared__ _Float16 htiles[WAVES_ * 16 * LAT_];                 // 16 KB
    __shared__ float    stages[DENSE > 0 ? WAVES_ * 16 * DENSE : 1];// <=32 KB
    const int wave = threadIdx.x >> 5;
    const int tile = blockIdx.x * WAVES_ + wave;
    mlp_block_tile<KPAD, DENSE>(afetch, dsrc, tile * 16, w0t, w1t, b0, b1,
                                cscale, cshift, resid, out,
                                htiles + wave * 16 * LAT_,
                                DENSE > 0 ? stages + wave * 16 * DENSE : stages);
}

// cond-LN MLP: h = silu(tau*w1+b1); scale = h@ws+bs; shift = h@wb+bb
__global__ void cond_kernel(const float* w1, const float* b1, const float* ws,
                            const float* bs, const float* wb, const float* bb,
                            const float* tau, float* out_scale, float* out_shift) {
    const int t = threadIdx.x;          // 128 threads
    const float tv = tau[0];
    float h[16];
#pragma unroll
    for (int j = 0; j < 16; ++j) h[j] = silu_f(tv * w1[j] + b1[j]);
    float sc = bs[t], sh = bb[t];
#pragma unroll
    for (int j = 0; j < 16; ++j) {
        sc += h[j] * ws[j * LAT_ + t];
        sh += h[j] * wb[j * LAT_ + t];
    }
    out_scale[t] = 1.0f + sc;
    out_shift[t] = sh;
}

// f32 [K,128] row-major -> f16 [128][Kpad] transposed (zero padded)
__global__ void wconv_kernel(const float* src, _Float16* dst, int kused,
                             int kpad, int krowoff) {
    const int i = blockIdx.x * 256 + threadIdx.x;
    if (i >= 128 * kpad) return;
    const int n = i / kpad;
    const int k = i % kpad;
    dst[i] = (k < kused) ? (_Float16)src[(size_t)(k + krowoff) * 128 + n]
                         : (_Float16)0.f;
}

__global__ void zero_kernel(float* p, int n) {
    const int i = blockIdx.x * 256 + threadIdx.x;
    if (i < n) p[i] = 0.f;
}

// segment sums + counts for segment-mean over receivers
__global__ void scatter_kernel(const float* elat, const int* recv,
                               float* sums, float* cnts) {
    const size_t i = (size_t)blockIdx.x * 256 + threadIdx.x;   // over B*E*128
    if (i >= (size_t)B_ * E_ * LAT_) return;
    const int c   = (int)(i & 127);
    const int row = (int)(i >> 7);         // b*E + e
    const int b   = row >> 16;
    const int r   = recv[row];
    atomicAdd(&sums[((size_t)(b << 11) + r) * LAT_ + c],
              elat[(size_t)row * LAT_ + c]);
    if (c == 0) atomicAdd(&cnts[(b << 11) + r], 1.0f);
}

// ---------------------------------------------------------------------------
// Host-side orchestration
// ---------------------------------------------------------------------------
extern "C" void kernel_launch(void* const* d_in, const int* in_sizes, int n_in,
                              void* d_out, int out_size, void* d_ws, size_t ws_size,
                              hipStream_t stream) {
    (void)in_sizes; (void)n_in; (void)out_size; (void)ws_size;

    // ---- inputs (recursive dict insertion order of setup_inputs) ----
    const float* pfeat = (const float*)d_in[0];
    const float* pstru = (const float*)d_in[1];
    const float* rstru = (const float*)d_in[2];
    const float* efeat = (const float*)d_in[3];
    const int*   send  = (const int*)d_in[4];
    const int*   recv  = (const int*)d_in[5];
    const float* tau   = (const float*)d_in[6];
    // params: pn_emb(7..10) pn_emb_n(11..16) rn_emb(17..20) rn_emb_n(21..26)
    //         e_emb(27..30) e_emb_n(31..36)  e_upd(37..40)  e_upd_n(41..46)
    //         rn_upd(47..50) rn_upd_n(51..56) pn_upd(57..60) pn_upd_n(61..66)
    // mlp order: W0, b0, W1, b1 ; cond order: w1, b1, ws, bs, wb, bb

    // ---- workspace layout ----
    size_t off = 0;
    float* cond = (float*)((char*)d_ws + off);          // 6 sets x (scale,shift)x128
    off += 6 * 2 * LAT_ * sizeof(float);
    off = (off + 255) & ~(size_t)255;

    _Float16* wh = (_Float16*)((char*)d_ws + off);
    size_t h = 0;
    _Float16* pn_w0t = wh + h; h += (size_t)128 * 96;
    _Float16* pn_w1t = wh + h; h += (size_t)128 * 128;
    _Float16* rn_w0t = wh + h; h += (size_t)128 * 32;
    _Float16* rn_w1t = wh + h; h += (size_t)128 * 128;
    _Float16* ee_w0t = wh + h; h += (size_t)128 * 32;
    _Float16* ee_w1t = wh + h; h += (size_t)128 * 128;
    _Float16* eu_w0t = wh + h; h += (size_t)128 * 384;
    _Float16* eu_w1t = wh + h; h += (size_t)128 * 128;
    _Float16* ru_w0t = wh + h; h += (size_t)128 * 256;
    _Float16* ru_w1t = wh + h; h += (size_t)128 * 128;
    _Float16* pu_w0t = wh + h; h += (size_t)128 * 128;
    _Float16* pu_w1t = wh + h; h += (size_t)128 * 128;
    off += h * sizeof(_Float16);
    off = (off + 255) & ~(size_t)255;

    float* sums = (float*)((char*)d_ws + off); off += (size_t)B_ * NR_ * LAT_ * 4;
    float* cnts = (float*)((char*)d_ws + off); off += (size_t)B_ * NR_ * 4;
    off = (off + 255) & ~(size_t)255;
    float* elat = (float*)((char*)d_ws + off); off += (size_t)B_ * E_ * LAT_ * 4;

    float* r_lat = (float*)d_out;                            // [B*NR,128]
    float* p_lat = (float*)d_out + (size_t)B_ * NR_ * LAT_;  // [B*NP,128]

    // ---- prep: cond vectors ----
    const int cidx[6] = {11, 21, 31, 41, 51, 61};            // w1 index per set
    for (int s = 0; s < 6; ++s) {
        const int i = cidx[s];
        cond_kernel<<<1, 128, 0, stream>>>(
            (const float*)d_in[i], (const float*)d_in[i + 1],
            (const float*)d_in[i + 2], (const float*)d_in[i + 3],
            (const float*)d_in[i + 4], (const float*)d_in[i + 5],
            tau, cond + s * 256, cond + s * 256 + 128);
    }

    // ---- prep: f16 transposed weights ----
    struct WJob { int src; _Float16* dst; int kused, kpad, kroff; };
    const WJob jobs[12] = {
        {7,  pn_w0t, 68, 96, 0},  {9,  pn_w1t, 128, 128, 0},
        {17, rn_w0t, 4,  32, 64}, {19, rn_w1t, 128, 128, 0},
        {27, ee_w0t, 3,  32, 0},  {29, ee_w1t, 128, 128, 0},
        {37, eu_w0t, 384, 384, 0},{39, eu_w1t, 128, 128, 0},
        {47, ru_w0t, 256, 256, 0},{49, ru_w1t, 128, 128, 0},
        {57, pu_w0t, 128, 128, 0},{59, pu_w1t, 128, 128, 0},
    };
    for (int j = 0; j < 12; ++j) {
        const int n = 128 * jobs[j].kpad;
        wconv_kernel<<<(n + 255) / 256, 256, 0, stream>>>(
            (const float*)d_in[jobs[j].src], jobs[j].dst,
            jobs[j].kused, jobs[j].kpad, jobs[j].kroff);
    }

    // ---- zero segment sums + counts (contiguous region) ----
    {
        const int n = B_ * NR_ * LAT_ + B_ * NR_;
        zero_kernel<<<(n + 255) / 256, 256, 0, stream>>>(sums, n);
    }

    const int TPB = 32 * WAVES_;          // 128 threads, 4 tiles per block
    const int RPB = 16 * WAVES_;          // 64 rows per block

    // ---- encoder embeds ----
    mlp_kernel<96, 0, PnEmbF><<<(B_ * NP_) / RPB, TPB, 0, stream>>>(
        PnEmbF{pfeat, pstru}, nullptr, pn_w0t, pn_w1t,
        (const float*)d_in[8], (const float*)d_in[10],
        cond + 0 * 256, cond + 0 * 256 + 128, nullptr, p_lat);

    mlp_kernel<32, 0, RnEmbF><<<(B_ * NR_) / RPB, TPB, 0, stream>>>(
        RnEmbF{rstru}, nullptr, rn_w0t, rn_w1t,
        (const float*)d_in[18], (const float*)d_in[20],
        cond + 1 * 256, cond + 1 * 256 + 128, nullptr, r_lat);

    mlp_kernel<32, 0, EEmbF><<<(B_ * E_) / RPB, TPB, 0, stream>>>(
        EEmbF{efeat}, nullptr, ee_w0t, ee_w1t,
        (const float*)d_in[28], (const float*)d_in[30],
        cond + 2 * 256, cond + 2 * 256 + 128, nullptr, elat);

    // ---- edge update (async-staged e_lat prefix + gathers + residual) ----
    mlp_kernel<384, 128, EUpdF><<<(B_ * E_) / RPB, TPB, 0, stream>>>(
        EUpdF{elat, p_lat, r_lat, send, recv}, elat, eu_w0t, eu_w1t,
        (const float*)d_in[38], (const float*)d_in[40],
        cond + 3 * 256, cond + 3 * 256 + 128, elat, elat);

    // ---- segment mean scatter ----
    {
        const size_t n = (size_t)B_ * E_ * LAT_;
        scatter_kernel<<<(unsigned)((n + 255) / 256), 256, 0, stream>>>(
            elat, recv, sums, cnts);
    }

    // ---- rnode update (async-staged r_lat prefix + residual) ----
    mlp_kernel<256, 128, RnUpdF><<<(B_ * NR_) / RPB, TPB, 0, stream>>>(
        RnUpdF{r_lat, sums, cnts}, r_lat, ru_w0t, ru_w1t,
        (const float*)d_in[48], (const float*)d_in[50],
        cond + 4 * 256, cond + 4 * 256 + 128, r_lat, r_lat);

    // ---- pnode update (fully async-staged + residual) ----
    mlp_kernel<128, 128, PnUpdF><<<(B_ * NP_) / RPB, TPB, 0, stream>>>(
        PnUpdF{p_lat}, p_lat, pu_w0t, pu_w1t,
        (const float*)d_in[58], (const float*)d_in[60],
        cond + 5 * 256, cond + 5 * 256 + 128, p_lat, p_lat);
}
